// SelfAttention_85907935854651
// MI455X (gfx1250) — matrix operand order repaired and checked by reading, not compiled, not run on hardware
//
#include <hip/hip_runtime.h>
#include <hip/hip_bf16.h>
#include <stdint.h>

typedef __attribute__((ext_vector_type(16))) _Float16 v16h;
typedef __attribute__((ext_vector_type(8)))  float    v8f;

#define BATCH 8
#define SEQ   2048
#define DM    1024
#define E     64
#define ROWS  (BATCH*SEQ)   // 16384

union Frag  { v16h h; uint32_t u[8]; };
union FragC { v8f  f; float x[8]; };

__device__ __forceinline__ uint32_t pack2f16(float x, float y) {
  union { _Float16 h[2]; uint32_t u; } t;
  t.h[0] = (_Float16)x; t.h[1] = (_Float16)y;
  return t.u;
}

// A-matrix 16x32 f16 layout (ISA 7.12.2): lanes 0-15: V0-3 = K0..7, V4-7 = K16..23;
// lanes 16-31: +8.
__device__ __forceinline__ int kA(int i, int hid) {
  return hid * 8 + (i < 4 ? 2 * i : 8 + 2 * i);
}
// B-matrix 32x16 f16, same K order as A: lanes 0-15: V0-3 = K0..7, V4-7 = K16..23;
// lanes 16-31: +8.
__device__ __forceinline__ int kB(int i, int hid) {
  return hid * 8 + (i < 4 ? 2 * i : 8 + 2 * i);
}

// ---------------------------------------------------------------------------
// Projection: out[m,e] = sum_d X[m,d] * W[e,d]   (M=16384, K=1024, N=64)
// f32 in -> f16 out. TRANSPOSE=true stores per-batch transposed [B,64,SEQ].
// Each wave: 16 rows x 32 cols (2 n-tiles) -> 2048 waves for latency hiding.
// transposeOut is a template param so the store epilogue is branch-free.
// ---------------------------------------------------------------------------
template <bool TRANSPOSE>
__global__ void proj_f16_kernel(const float* __restrict__ X,
                                const float* __restrict__ W,
                                _Float16* __restrict__ out) {
  const int lane = threadIdx.x & 31;
  const int wave = threadIdx.x >> 5;
  const int r    = lane & 15;
  const int hid  = lane >> 4;
  const int wl   = blockIdx.x * 8 + wave;  // 0..2047
  const int nh   = wl & 1;                 // which pair of n-tiles
  const int tileM = (wl >> 1) * 16;

  const float* arow  = X + (size_t)(tileM + r) * DM;
  const float* brow0 = W + (size_t)((nh * 2 + 0) * 16 + r) * DM;
  const float* brow1 = W + (size_t)((nh * 2 + 1) * 16 + r) * DM;

  v8f acc0 = {}, acc1 = {};

  for (int kb = 0; kb < DM; kb += 32) {
    __builtin_prefetch(arow + kb + 64, 0, 3);
    Frag a;
#pragma unroll
    for (int i = 0; i < 8; ++i) {
      int k = kb + kA(i, hid);
      a.u[i] = pack2f16(arow[k], arow[k + 1]);
    }
    Frag b0, b1;
#pragma unroll
    for (int i = 0; i < 8; ++i) {
      int k = kb + kB(i, hid);
      b0.u[i] = pack2f16(brow0[k], brow0[k + 1]);
      b1.u[i] = pack2f16(brow1[k], brow1[k + 1]);
    }
    acc0 = __builtin_amdgcn_wmma_f32_16x16x32_f16(false, a.h, false, b0.h, (short)0, acc0, false, false);
    acc1 = __builtin_amdgcn_wmma_f32_16x16x32_f16(false, a.h, false, b1.h, (short)0, acc1, false, false);
  }

  v8f accs[2] = {acc0, acc1};
#pragma unroll
  for (int t = 0; t < 2; ++t) {
#pragma unroll
    for (int j = 0; j < 8; ++j) {
      int row = tileM + j + hid * 8;
      int col = (nh * 2 + t) * 16 + r;
      _Float16 val = (_Float16)accs[t][j];
      if (TRANSPOSE) {
        int bb = row >> 11;          // row / SEQ
        int ss = row & (SEQ - 1);
        out[((size_t)(bb * E + col)) * SEQ + ss] = val;
      } else {
        out[(size_t)row * E + col] = val;
      }
    }
  }
}

// ---------------------------------------------------------------------------
// Fused scores + softmax. One 512-thread block (16 waves) per 16-row query
// tile: wave w owns 128 key columns (8 col tiles, 16 WMMAs), full row of
// 2048 scores lives in block registers. Row max/sum via shfl within the
// 16-lane half-wave + LDS across the 16 waves. Writes probs f32 (d_out) and
// probs f16 (workspace) exactly once.
// ---------------------------------------------------------------------------
__global__ __launch_bounds__(512) void scores_softmax_kernel(
    const _Float16* __restrict__ qp, const _Float16* __restrict__ kp,
    float* __restrict__ probs, _Float16* __restrict__ probs16) {
  __shared__ float redm[16][16];   // [wave][row-in-tile]
  __shared__ float rowred[16];
  const int tid  = threadIdx.x;
  const int lane = tid & 31;
  const int wave = tid >> 5;        // 0..15
  const int r    = lane & 15;
  const int hid  = lane >> 4;
  const int b    = blockIdx.x >> 7; // gridDim.x = 8*128
  const int rt   = blockIdx.x & 127;

  const _Float16* aptr = qp + ((size_t)(b * SEQ + rt * 16 + r)) * E;
  Frag a0, a1;
#pragma unroll
  for (int i = 0; i < 8; ++i) {
    a0.u[i] = *(const uint32_t*)(aptr + kA(i, hid));
    a1.u[i] = *(const uint32_t*)(aptr + 32 + kA(i, hid));
  }

  FragC accs[8];
#pragma unroll
  for (int ct = 0; ct < 8; ++ct) {
    int colb = wave * 128 + ct * 16;
    const _Float16* bptr = kp + ((size_t)(b * SEQ + colb + r)) * E;
    Frag b0, b1;
#pragma unroll
    for (int i = 0; i < 8; ++i) {
      b0.u[i] = *(const uint32_t*)(bptr + kB(i, hid));
      b1.u[i] = *(const uint32_t*)(bptr + 32 + kB(i, hid));
    }
    v8f acc = {};
    acc = __builtin_amdgcn_wmma_f32_16x16x32_f16(false, a0.h, false, b0.h, (short)0, acc, false, false);
    acc = __builtin_amdgcn_wmma_f32_16x16x32_f16(false, a1.h, false, b1.h, (short)0, acc, false, false);
    accs[ct].f = acc;
  }

  const float inv = 1.0f / 64.0f;

  // ---- row max (rows: j + hid*8) ----
  float m[8];
#pragma unroll
  for (int j = 0; j < 8; ++j) {
    float x = accs[0].x[j];
#pragma unroll
    for (int ct = 1; ct < 8; ++ct) x = fmaxf(x, accs[ct].x[j]);
#pragma unroll
    for (int o = 1; o < 16; o <<= 1) x = fmaxf(x, __shfl_xor(x, o, 32));
    m[j] = x * inv;
  }
  if (r == 0) {
#pragma unroll
    for (int j = 0; j < 8; ++j) redm[wave][hid * 8 + j] = m[j];
  }
  __syncthreads();
  if (tid < 16) {
    float x = redm[0][tid];
#pragma unroll
    for (int w = 1; w < 16; ++w) x = fmaxf(x, redm[w][tid]);
    rowred[tid] = x;
  }
  __syncthreads();
  float rm[8];
#pragma unroll
  for (int j = 0; j < 8; ++j) rm[j] = rowred[hid * 8 + j];

  // ---- exp + row sum ----
  float s[8];
#pragma unroll
  for (int j = 0; j < 8; ++j) s[j] = 0.0f;
#pragma unroll
  for (int ct = 0; ct < 8; ++ct) {
#pragma unroll
    for (int j = 0; j < 8; ++j) {
      float e = __expf(accs[ct].x[j] * inv - rm[j]);
      accs[ct].x[j] = e;
      s[j] += e;
    }
  }
#pragma unroll
  for (int j = 0; j < 8; ++j) {
#pragma unroll
    for (int o = 1; o < 16; o <<= 1) s[j] += __shfl_xor(s[j], o, 32);
  }
  if (r == 0) {
#pragma unroll
    for (int j = 0; j < 8; ++j) redm[wave][hid * 8 + j] = s[j];
  }
  __syncthreads();
  if (tid < 16) {
    float x = 0.0f;
#pragma unroll
    for (int w = 0; w < 16; ++w) x += redm[w][tid];
    rowred[tid] = x;
  }
  __syncthreads();
  float rs[8];
#pragma unroll
  for (int j = 0; j < 8; ++j) rs[j] = 1.0f / rowred[hid * 8 + j];

  // ---- store normalized probs: f32 to d_out, f16 to workspace ----
  size_t pbase = (size_t)b * SEQ * SEQ;
#pragma unroll
  for (int ct = 0; ct < 8; ++ct) {
    int col = wave * 128 + ct * 16 + r;
#pragma unroll
    for (int j = 0; j < 8; ++j) {
      int row = rt * 16 + j + hid * 8;
      float p = accs[ct].x[j] * rs[j];
      size_t idx = pbase + (size_t)row * SEQ + col;
      probs[idx]   = p;
      probs16[idx] = (_Float16)p;
    }
  }
}

// ---------------------------------------------------------------------------
// att[b,q,e] = sum_k probs16[b,q,k] * vp[b,k,e]   (K=2048), f16 probs input.
// Each wave: one 16x16 tile -> 4096 waves (occupancy for the 67MB stream);
// A re-reads stay in the 192MB L2 (probs16+vpT = 66MB resident).
// ---------------------------------------------------------------------------
__global__ void att16_kernel(const _Float16* __restrict__ probs16,
                             const _Float16* __restrict__ vpT,
                             float* __restrict__ att) {
  const int lane = threadIdx.x & 31;
  const int wave = threadIdx.x >> 5;
  const int r    = lane & 15;
  const int hid  = lane >> 4;
  int wl = blockIdx.x * 8 + wave;      // 0..4095
  int nt = wl & 3;
  int rt = (wl >> 2) & 127;
  int b  = wl >> 9;

  const _Float16* aptr = probs16 + (size_t)b * SEQ * SEQ + (size_t)(rt * 16 + r) * SEQ;
  const _Float16* bp   = vpT + ((size_t)(b * E + nt * 16 + r)) * SEQ;

  v8f acc = {};

#pragma unroll 4
  for (int kb = 0; kb < SEQ; kb += 32) {
    __builtin_prefetch(aptr + kb + 256, 0, 3);
    Frag a, bfr;
#pragma unroll
    for (int i = 0; i < 8; ++i) {
      a.u[i]   = *(const uint32_t*)(aptr + kb + kA(i, hid));
      bfr.u[i] = *(const uint32_t*)(bp + kb + kB(i, hid));
    }
    acc = __builtin_amdgcn_wmma_f32_16x16x32_f16(false, a.h, false, bfr.h, (short)0, acc, false, false);
  }

#pragma unroll
  for (int j = 0; j < 8; ++j) {
    int row = rt * 16 + j + hid * 8;
    att[((size_t)(b * SEQ + row)) * E + nt * 16 + r] = acc[j];
  }
}

// ===========================================================================
// Fallback pipeline (used only if ws_size can't hold the f16 probs shadow)
// ===========================================================================
__global__ void scores_kernel(const _Float16* __restrict__ qp,
                              const _Float16* __restrict__ kp,
                              float* __restrict__ probs) {
  const int lane = threadIdx.x & 31;
  const int wave = threadIdx.x >> 5;
  const int r    = lane & 15;
  const int hid  = lane >> 4;
  int wl = blockIdx.x * 8 + wave;
  int cg = wl & 31;
  int rt = (wl >> 5) & 127;
  int b  = wl >> 12;

  const _Float16* aptr = qp + ((size_t)(b * SEQ + rt * 16 + r)) * E;
  Frag a0, a1;
#pragma unroll
  for (int i = 0; i < 8; ++i) {
    a0.u[i] = *(const uint32_t*)(aptr + kA(i, hid));
    a1.u[i] = *(const uint32_t*)(aptr + 32 + kA(i, hid));
  }
  const float inv = 1.0f / 64.0f;
  size_t pbase = (size_t)b * SEQ * SEQ;
#pragma unroll
  for (int ct = 0; ct < 4; ++ct) {
    int colb = cg * 64 + ct * 16;
    const _Float16* bptr = kp + ((size_t)(b * SEQ + colb + r)) * E;
    Frag b0, b1;
#pragma unroll
    for (int i = 0; i < 8; ++i) {
      b0.u[i] = *(const uint32_t*)(bptr + kB(i, hid));
      b1.u[i] = *(const uint32_t*)(bptr + 32 + kB(i, hid));
    }
    v8f acc = {};
    acc = __builtin_amdgcn_wmma_f32_16x16x32_f16(false, a0.h, false, b0.h, (short)0, acc, false, false);
    acc = __builtin_amdgcn_wmma_f32_16x16x32_f16(false, a1.h, false, b1.h, (short)0, acc, false, false);
#pragma unroll
    for (int j = 0; j < 8; ++j) {
      int row = rt * 16 + j + hid * 8;
      probs[pbase + (size_t)row * SEQ + colb + r] = acc[j] * inv;
    }
  }
}

__global__ void softmax_kernel(float* __restrict__ probs) {
  __shared__ float sred[8];
  const int tid = threadIdx.x;
  float* row = probs + (size_t)blockIdx.x * SEQ;
  float v[8];
#pragma unroll
  for (int i = 0; i < 8; ++i) v[i] = row[tid + i * 256];
  float m = v[0];
#pragma unroll
  for (int i = 1; i < 8; ++i) m = fmaxf(m, v[i]);
#pragma unroll
  for (int o = 16; o > 0; o >>= 1) m = fmaxf(m, __shfl_xor(m, o, 32));
  if ((tid & 31) == 0) sred[tid >> 5] = m;
  __syncthreads();
  if (tid < 32) {
    float x = (tid < 8) ? sred[tid] : -3.4e38f;
#pragma unroll
    for (int o = 4; o > 0; o >>= 1) x = fmaxf(x, __shfl_xor(x, o, 32));
    if (tid == 0) sred[0] = x;
  }
  __syncthreads();
  m = sred[0];
  __syncthreads();
  float s = 0.0f;
#pragma unroll
  for (int i = 0; i < 8; ++i) { v[i] = __expf(v[i] - m); s += v[i]; }
#pragma unroll
  for (int o = 16; o > 0; o >>= 1) s += __shfl_xor(s, o, 32);
  if ((tid & 31) == 0) sred[tid >> 5] = s;
  __syncthreads();
  if (tid < 32) {
    float x = (tid < 8) ? sred[tid] : 0.0f;
#pragma unroll
    for (int o = 4; o > 0; o >>= 1) x += __shfl_xor(x, o, 32);
    if (tid == 0) sred[0] = x;
  }
  __syncthreads();
  float rinv = 1.0f / sred[0];
#pragma unroll
  for (int i = 0; i < 8; ++i) row[tid + i * 256] = v[i] * rinv;
}

__global__ void att_f32_kernel(const float* __restrict__ probs,
                               const _Float16* __restrict__ vpT,
                               float* __restrict__ att) {
  const int lane = threadIdx.x & 31;
  const int wave = threadIdx.x >> 5;
  const int r    = lane & 15;
  const int hid  = lane >> 4;
  int wl = blockIdx.x * 8 + wave;      // 0..4095
  int nt = wl & 3;
  int rt = (wl >> 2) & 127;
  int b  = wl >> 9;

  const float* aptr = probs + (size_t)b * SEQ * SEQ + (size_t)(rt * 16 + r) * SEQ;
  const _Float16* bp = vpT + ((size_t)(b * E + nt * 16 + r)) * SEQ;

  v8f acc = {};
  for (int kb = 0; kb < SEQ; kb += 32) {
    Frag a, bfr;
#pragma unroll
    for (int i = 0; i < 8; ++i) {
      int k = kb + kA(i, hid);
      a.u[i]   = pack2f16(aptr[k], aptr[k + 1]);
      bfr.u[i] = *(const uint32_t*)(bp + kb + kB(i, hid));
    }
    acc = __builtin_amdgcn_wmma_f32_16x16x32_f16(false, a.h, false, bfr.h, (short)0, acc, false, false);
  }
#pragma unroll
  for (int j = 0; j < 8; ++j) {
    int row = rt * 16 + j + hid * 8;
    att[((size_t)(b * SEQ + row)) * E + nt * 16 + r] = acc[j];
  }
}

// ---------------------------------------------------------------------------
extern "C" void kernel_launch(void* const* d_in, const int* in_sizes, int n_in,
                              void* d_out, int out_size, void* d_ws, size_t ws_size,
                              hipStream_t stream) {
  const float* q  = (const float*)d_in[0];
  const float* k  = (const float*)d_in[1];
  const float* v  = (const float*)d_in[2];
  const float* Wq = (const float*)d_in[3];
  const float* Wk = (const float*)d_in[4];
  const float* Wv = (const float*)d_in[5];

  float* att   = (float*)d_out;                           // [B,S,64]
  float* probs = (float*)d_out + (size_t)ROWS * E;        // [B,S,S]

  _Float16* qp  = (_Float16*)d_ws;                        // [B,S,64] f16
  _Float16* kp  = qp + (size_t)ROWS * E;                  // [B,S,64] f16
  _Float16* vpT = kp + (size_t)ROWS * E;                  // [B,64,S] f16
  _Float16* probs16 = vpT + (size_t)ROWS * E;             // [B,S,S]  f16

  const size_t projBytes = (size_t)3 * ROWS * E * sizeof(_Float16);
  const size_t p16Bytes  = (size_t)BATCH * SEQ * SEQ * sizeof(_Float16);

  proj_f16_kernel<false><<<256, 256, 0, stream>>>(q, Wq, qp);
  proj_f16_kernel<false><<<256, 256, 0, stream>>>(k, Wk, kp);
  proj_f16_kernel<true ><<<256, 256, 0, stream>>>(v, Wv, vpT);

  if (ws_size >= projBytes + p16Bytes) {
    // Fast path: fused scores+softmax (writes probs once, f32 + f16 shadow),
    // then f16-only PV GEMM.
    scores_softmax_kernel<<<BATCH * 128, 512, 0, stream>>>(qp, kp, probs, probs16);
    att16_kernel<<<512, 256, 0, stream>>>(probs16, vpT, att);
  } else {
    scores_kernel<<<4096, 256, 0, stream>>>(qp, kp, probs);
    softmax_kernel<<<16384, 256, 0, stream>>>(probs);
    att_f32_kernel<<<512, 256, 0, stream>>>(probs, vpT, att);
  }
}
